// NequIP_34557306863891
// MI455X (gfx1250) — matrix operand order repaired and checked by reading, not compiled, not run on hardware
//
#include <hip/hip_runtime.h>
#include <math.h>

// ---------------------------------------------------------------------------
// Types / constants
// ---------------------------------------------------------------------------
typedef __attribute__((ext_vector_type(16))) __bf16 v16bf;
typedef __attribute__((ext_vector_type(8)))  float  v8f;

#define NSP 118
#define CGTOT (23*343)
#define RCf 5.0f
#define PIF 3.14159265358979323846f
#define INV_DENOM 0.05f

constexpr int PL1[23] = {0,0,0,0, 1,1,1,1,1,1, 2,2,2,2,2,2,2, 3,3,3,3,3,3};
constexpr int PL2[23] = {0,1,2,3, 0,1,1,2,2,3, 0,1,1,2,2,3,3, 0,1,2,2,3,3};
constexpr int PL3[23] = {0,1,2,3, 1,0,2,1,3,2, 2,1,3,0,2,1,3, 3,2,1,3,0,2};
constexpr int AOFF[4] = {0,1,4,9};

struct LTab { int np; int pidx[23]; };
constexpr LTab LTABS[3] = {
  {4 ,{0,1,2,3}},
  {23,{0,1,2,3,4,5,6,7,8,9,10,11,12,13,14,15,16,17,18,19,20,21,22}},
  {4 ,{0,5,13,21}}
};
constexpr unsigned L3MASK[3] = {0xFFFFu,0xFFFFu,0x0001u};
constexpr unsigned L1MASK[3] = {0x0001u,0xFFFFu,0xFFFFu};

__device__ __forceinline__ float siluf(float x){ float s=1.f/(1.f+__expf(-x)); return x*s; }
__device__ __forceinline__ float silugrad(float x){ float s=1.f/(1.f+__expf(-x)); return s*(1.f+x*(1.f-s)); }

// ---------------------------------------------------------------------------
// CG coefficient generation (Racah formula + real spherical harmonic basis)
// ---------------------------------------------------------------------------
struct dcplx { double re, im; };

__device__ double dfact(int n){ double r=1.0; for(int i=2;i<=n;++i) r*=(double)i; return r; }

__device__ double cg_coef(int j1,int m1,int j2,int m2,int j3,int m3){
  if (m1+m2!=m3) return 0.0;
  if (m1< -j1||m1>j1||m2<-j2||m2>j2||m3<-j3||m3>j3) return 0.0;
  double pref = sqrt((2.0*j3+1.0)*dfact(j3+j1-j2)*dfact(j3-j1+j2)*dfact(j1+j2-j3)/dfact(j1+j2+j3+1));
  pref *= sqrt(dfact(j3+m3)*dfact(j3-m3)*dfact(j1-m1)*dfact(j1+m1)*dfact(j2-m2)*dfact(j2+m2));
  double s=0.0;
  for(int k=0;k<=j1+j2-j3;++k){
    int d0=k, d1=j1+j2-j3-k, d2=j1-m1-k, d3=j2+m2-k, d4=j3-j2+m1+k, d5=j3-j1-m2+k;
    if(d0<0||d1<0||d2<0||d3<0||d4<0||d5<0) continue;
    double den = dfact(d0)*dfact(d1)*dfact(d2)*dfact(d3)*dfact(d4)*dfact(d5);
    s += ((k&1)? -1.0:1.0)/den;
  }
  return pref*s;
}

__device__ dcplx u_entry(int l,int x,int y){
  int m=x-l, mp=y-l;
  const double inv=0.7071067811865476;
  dcplx zz; zz.re=0.0; zz.im=0.0;
  if(m==0){ if(mp==0) zz.re=1.0; }
  else if(m>0){
    if(mp==-m) zz.re=inv;
    else if(mp==m) zz.re=((m&1)?-1.0:1.0)*inv;
  } else {
    int am=-m;
    if(mp==m) zz.im=inv;
    else if(mp==-m) zz.im = -((am&1)?-1.0:1.0)*inv;
  }
  return zz;
}

__global__ void cg_init_kernel(float* __restrict__ cg){
  int p = blockIdx.x;
  int t = threadIdx.x;               // 0..342
  int l1=PL1[p], l2=PL2[p], l3=PL3[p];
  int x=t/49, y=(t/7)%7, zc=t%7;
  float val=0.f;
  if(x<2*l1+1 && y<2*l2+1 && zc<2*l3+1){
    double accr=0.0;
    for(int a=0;a<2*l1+1;++a){
      dcplx u1=u_entry(l1,x,a);
      if(u1.re==0.0 && u1.im==0.0) continue;
      for(int b=0;b<2*l2+1;++b){
        dcplx u2=u_entry(l2,y,b);
        if(u2.re==0.0 && u2.im==0.0) continue;
        double pre=u1.re*u2.re-u1.im*u2.im;
        double pim=u1.re*u2.im+u1.im*u2.re;
        for(int c=0;c<2*l3+1;++c){
          dcplx u3=u_entry(l3,zc,c);
          if(u3.re==0.0 && u3.im==0.0) continue;
          double cc=cg_coef(l1,a-l1,l2,b-l2,l3,c-l3);
          if(cc==0.0) continue;
          accr += (pre*u3.re + pim*u3.im)*cc;   // times conj(u3)
        }
      }
    }
    val=(float)accr;
  }
  cg[p*343 + x*49 + y*7 + zc] = val;
}

// ---------------------------------------------------------------------------
// Weight conversion helpers (f32 -> bf16, plus transposed / padded variants)
// ---------------------------------------------------------------------------
__global__ void cvt_kernel(const float* __restrict__ s, __bf16* __restrict__ d, int n){
  int i=blockIdx.x*256+threadIdx.x; if(i<n) d[i]=(__bf16)s[i];
}
__global__ void cvt_pad_rows_kernel(const float* __restrict__ s, __bf16* __restrict__ d, int R,int C,int Rp){
  int i=blockIdx.x*256+threadIdx.x;
  if(i<Rp*C){ int r=i/C,c=i%C; d[i]= (r<R)?(__bf16)s[r*C+c]:(__bf16)0.f; }
}
__global__ void cvt_T_pad_kernel(const float* __restrict__ s, __bf16* __restrict__ d, int R,int C,int Rp){
  int i=blockIdx.x*256+threadIdx.x;
  if(i<C*Rp){ int c=i/Rp, r=i%Rp; d[i]=(r<R)?(__bf16)s[r*C+c]:(__bf16)0.f; }
}
__global__ void vout_kernel(const float* __restrict__ W1,const float* __restrict__ W2,float* __restrict__ vout){
  int c=threadIdx.x;
  if(c<32){ float s=0.f; for(int k=0;k<16;++k) s+=W1[c*16+k]*W2[k]; vout[c]=s; }
}

// ---------------------------------------------------------------------------
// WMMA bf16 GEMM : out[M,N](f32) = A[M,K](bf16,row-major) * B[K,N](bf16,ldb)
// FUSE bits: 1=silu  2=write preF(f32)  4=write outF(f32)  8=write outH(bf16)
//            16=accumulate into outF
//
// Block owns ONE 16-wide N-tile (blockIdx.y) and 8 waves x MT M-tiles
// (blockIdx.x). The K x 16 B-panel is gathered once into LDS, pre-swizzled
// into a per-lane B-fragment layout (k = 16*(lane/16) + j, same as A), so the inner
// loop reads B with two ds_load_b128 instead of 16 scalar global loads.
// Each staged fragment feeds MT WMMA ops. All tile predicates are
// wave-uniform -> EXEC stays all-ones at every WMMA.
// ---------------------------------------------------------------------------
#define GEMM_MT 4
#define GEMM_MAXKB 23   // max K/32 (K<=736)

template<int FUSE>
__global__ __launch_bounds__(256) void gemm_kernel(
    const __bf16* __restrict__ A, const __bf16* __restrict__ B,
    float* __restrict__ outF, __bf16* __restrict__ outH, float* __restrict__ preF,
    int M, int N, int K, int ldb)
{
  __shared__ __align__(16) __bf16 Bs[GEMM_MAXKB][32][16];
  const int tid = threadIdx.x;
  const int col0 = blockIdx.y*16;
  const int nkb = K>>5;
  // ---- stage + swizzle B panel: Bs[kb][lane][j] = B[(kb*32 + khalf*16 + j)*ldb + col0 + (lane&15)]
  for(int i=tid;i<nkb*32;i+=256){
    int kb=i>>5, lane=i&31;
    int krow = kb*32 + ((lane>>4)<<4);
    int cc = col0 + (lane&15);
    #pragma unroll
    for(int j=0;j<16;++j) Bs[kb][lane][j] = B[(size_t)(krow+j)*ldb + cc];
  }
  __syncthreads();
  const int wave=tid>>5, lane=tid&31;
  const long long tM0 = ((long long)blockIdx.x*8 + wave)*GEMM_MT;  // first M-tile
  const int ka=(lane>>4)*16;
  v8f zero = {};
  v8f acc[GEMM_MT];
  #pragma unroll
  for(int m=0;m<GEMM_MT;++m) acc[m]=zero;
  for(int kb=0;kb<nkb;++kb){
    v16bf b;
    const __bf16* bp=&Bs[kb][lane][0];
    #pragma unroll
    for(int j=0;j<16;++j) b[j]=bp[j];
    #pragma unroll
    for(int m=0;m<GEMM_MT;++m){
      long long tM=tM0+m;
      if(tM*16>=M) break;                       // wave-uniform
      int row=(int)(tM*16)+(lane&15); if(row>M-1) row=M-1;
      const __bf16* Ap=A+(size_t)row*K + kb*32 + ka;
      v16bf a;
      #pragma unroll
      for(int j=0;j<16;++j){ a[j]=Ap[j]; }
      acc[m]=__builtin_amdgcn_wmma_f32_16x16x32_bf16(false,a,false,b,(short)0,acc[m],false,false);
    }
  }
  const int col = col0 + (lane&15);
  #pragma unroll
  for(int m=0;m<GEMM_MT;++m){
    long long tM=tM0+m;
    if(tM*16>=M) break;
    int rbase=(int)(tM*16)+(lane>>4)*8;
    #pragma unroll
    for(int i=0;i<8;++i){
      int r=rbase+i;
      if(r>=M) continue;
      size_t idx=(size_t)r*N+col;
      float v=acc[m][i];
      if(FUSE&2)  preF[idx]=v;
      if(FUSE&1)  v=siluf(v);
      if(FUSE&4)  outF[idx]=v;
      if(FUSE&16) outF[idx]+=v;
      if(FUSE&8)  outH[idx]=(__bf16)v;
    }
  }
}

// ---------------------------------------------------------------------------
// Edge geometry: r, u, real spherical harmonics (l<=3), Bessel radial basis
// ---------------------------------------------------------------------------
__global__ void edge_geom_kernel(const float* __restrict__ ev, float* __restrict__ sh,
                                 __bf16* __restrict__ rbf, int E){
  int e=blockIdx.x*blockDim.x+threadIdx.x;
  if(e>=E) return;
  float x=ev[e*3],y=ev[e*3+1],z=ev[e*3+2];
  float r=sqrtf(x*x+y*y+z*z);
  float rm=fmaxf(r,1e-9f);
  float ux=x/rm, uy=y/rm, uz=z/rm;
  const float s3=1.7320508075688772f, s5=2.23606797749979f, s15=3.872983346207417f;
  const float s7=2.6457513110645907f, s42=6.480740698407860f, s105=10.246950765959598f, s70=8.366600265340756f;
  float* sp=sh+(size_t)e*16;
  sp[0]=1.f;
  sp[1]=s3*uy; sp[2]=s3*uz; sp[3]=s3*ux;
  sp[4]=s15*ux*uy; sp[5]=s15*uy*uz; sp[6]=0.5f*s5*(3.f*uz*uz-1.f);
  sp[7]=s15*ux*uz; sp[8]=0.5f*s15*(ux*ux-uy*uy);
  sp[9] =0.25f*s70*uy*(3.f*ux*ux-uy*uy);
  sp[10]=s105*ux*uy*uz;
  sp[11]=0.25f*s42*uy*(5.f*uz*uz-1.f);
  sp[12]=0.5f*s7*(5.f*uz*uz*uz-3.f*uz);
  sp[13]=0.25f*s42*ux*(5.f*uz*uz-1.f);
  sp[14]=0.5f*s105*uz*(ux*ux-uy*uy);
  sp[15]=0.25f*s70*ux*(ux*ux-3.f*uy*uy);
  float xr=r/RCf;
  float env=0.f;
  if(xr<1.f){
    float x2=xr*xr,x3=x2*xr,x6=x3*x3,x7=x6*xr,x8=x7*xr;
    env=1.f-28.f*x6+48.f*x7-21.f*x8;
  }
  float S=sqrtf(2.f/RCf);
  __bf16* rp=rbf+(size_t)e*32;
  #pragma unroll
  for(int k=1;k<=8;++k) rp[k-1]=(__bf16)(S*sinf((float)k*PIF*xr)/rm*env);
  #pragma unroll
  for(int k=8;k<32;++k) rp[k]=(__bf16)0.f;
}

// ---------------------------------------------------------------------------
// Embedding / head
// ---------------------------------------------------------------------------
__global__ void embed_kernel(const float* __restrict__ Wemb, const int* __restrict__ Z,
                             float* __restrict__ X0, int NN){
  int i=blockIdx.x*256+threadIdx.x;
  if(i<NN*32){ int n=i>>5,c=i&31; X0[(size_t)n*512+c*16]=Wemb[(size_t)Z[n]*32+c]; }
}
__global__ void head_kernel(const float* __restrict__ Xf, const float* __restrict__ vout,
                            float* __restrict__ out, int NN){
  int n=blockIdx.x*blockDim.x+threadIdx.x;
  float ea=0.f;
  if(n<NN){
    const float* x=Xf+(size_t)n*512;
    #pragma unroll
    for(int c=0;c<32;++c) ea += x[c*16]*vout[c];
  }
  #pragma unroll
  for(int off=16;off>0;off>>=1) ea+=__shfl_xor(ea,off,32);
  if((threadIdx.x&31)==0) atomicAdd(&out[0], ea);
}
__global__ void init_head_grad_kernel(const float* __restrict__ vout, float* __restrict__ gX, int NN){
  int i=blockIdx.x*256+threadIdx.x;
  if(i<NN*32){ int n=i>>5,c=i&31; gX[(size_t)n*512+c*16]=vout[c]; }
}

// ---------------------------------------------------------------------------
// Tensor-product message passing (forward). Wave per edge, lane = channel c.
// ---------------------------------------------------------------------------
template<int T>
__global__ __launch_bounds__(256) void tp_forward(
  const float* __restrict__ Xin, const float* __restrict__ sh,
  const __bf16* __restrict__ w, const float* __restrict__ cg,
  const int* __restrict__ esrc, const int* __restrict__ edst,
  float* __restrict__ agg, int E)
{
  __shared__ float s_cg[CGTOT];
  for(int i=threadIdx.x;i<CGTOT;i+=256) s_cg[i]=cg[i];
  __syncthreads();
  int ln=threadIdx.x>>5, lane=threadIdx.x&31;
  int e=blockIdx.x*8+ln;
  if(e>=E) return;
  int src=esrc[e], dst=edst[e];
  float xs[16],shr[16],msg[16];
  const float* xp=Xin+(size_t)src*512+lane*16;
  #pragma unroll
  for(int k=0;k<16;++k){ xs[k]=xp[k]; shr[k]=sh[(size_t)e*16+k]; msg[k]=0.f; }
  constexpr int NP=LTABS[T].np;
  #pragma unroll
  for(int pi=0;pi<NP;++pi){
    const int p=LTABS[T].pidx[pi];
    const int l1=PL1[p],l2=PL2[p],l3=PL3[p];
    const int na=2*l1+1,nb=2*l2+1,nd=2*l3+1;
    const int ao=AOFF[l1],bo=AOFF[l2],dofs=AOFF[l3];
    float wv=(float)w[(size_t)e*(NP*32)+pi*32+lane];
    #pragma unroll
    for(int dd=0;dd<nd;++dd){
      float td=0.f;
      #pragma unroll
      for(int a=0;a<na;++a){
        float s=0.f;
        #pragma unroll
        for(int b=0;b<nb;++b) s+=shr[bo+b]*s_cg[p*343+a*49+b*7+dd];
        td+=xs[ao+a]*s;
      }
      msg[dofs+dd]+=wv*td;
    }
  }
  float* ap=agg+(size_t)dst*512+lane*16;
  #pragma unroll
  for(int k=0;k<16;++k)
    if((L3MASK[T]>>k)&1u) atomicAdd(&ap[k], msg[k]*INV_DENOM);
}

// ---------------------------------------------------------------------------
// TP backward: produces g_w (in place over w), scatters g_x to src nodes,
// accumulates g_sh per edge (cross-lane reduce over channels).
// ---------------------------------------------------------------------------
template<int T>
__global__ __launch_bounds__(256) void tp_backward(
  const float* __restrict__ Xin, const float* __restrict__ sh,
  const __bf16* w, const float* __restrict__ cg,
  const int* __restrict__ esrc, const int* __restrict__ edst,
  const float* __restrict__ gAgg,
  __bf16* gw, float* gXnext, float* __restrict__ gshT, int E)
{
  __shared__ float s_cg[CGTOT];
  for(int i=threadIdx.x;i<CGTOT;i+=256) s_cg[i]=cg[i];
  __syncthreads();
  int ln=threadIdx.x>>5, lane=threadIdx.x&31;
  int e=blockIdx.x*8+ln;
  if(e>=E) return;
  int src=esrc[e], dst=edst[e];
  float xs[16],shr[16],gm[16],gsh[16],gx[16];
  const float* xp=Xin+(size_t)src*512+lane*16;
  const float* gp=gAgg+(size_t)dst*512+lane*16;
  #pragma unroll
  for(int k=0;k<16;++k){
    xs[k]=xp[k]; shr[k]=sh[(size_t)e*16+k];
    gm[k]=((L3MASK[T]>>k)&1u)? gp[k]*INV_DENOM : 0.f;
    gsh[k]=0.f; gx[k]=0.f;
  }
  constexpr int NP=LTABS[T].np;
  #pragma unroll
  for(int pi=0;pi<NP;++pi){
    const int p=LTABS[T].pidx[pi];
    const int l1=PL1[p],l2=PL2[p],l3=PL3[p];
    const int na=2*l1+1,nb=2*l2+1,nd=2*l3+1;
    const int ao=AOFF[l1],bo=AOFF[l2],dofs=AOFF[l3];
    float wv=(float)w[(size_t)e*(NP*32)+pi*32+lane];
    float gwv=0.f;
    #pragma unroll
    for(int a=0;a<na;++a){
      float ta=0.f;
      #pragma unroll
      for(int b=0;b<nb;++b){
        float G=0.f;
        #pragma unroll
        for(int dd=0;dd<nd;++dd) G += gm[dofs+dd]*s_cg[p*343+a*49+b*7+dd];
        ta += shr[bo+b]*G;
        gsh[bo+b] += wv*xs[ao+a]*G;
      }
      gx[ao+a] += wv*ta;
      gwv += xs[ao+a]*ta;
    }
    gw[(size_t)e*(NP*32)+pi*32+lane]=(__bf16)gwv;
  }
  float* gxp=gXnext+(size_t)src*512+lane*16;
  #pragma unroll
  for(int k=0;k<16;++k)
    if((L1MASK[T]>>k)&1u) atomicAdd(&gxp[k], gx[k]);
  #pragma unroll
  for(int k=0;k<16;++k){
    float v=gsh[k];
    #pragma unroll
    for(int off=16;off>0;off>>=1) v+=__shfl_xor(v,off,32);
    if(lane==0) gshT[(size_t)e*16+k]+=v;
  }
}

// ---------------------------------------------------------------------------
// Node update forward. Wave per node, lane = output channel d.
// ---------------------------------------------------------------------------
template<int T>
__global__ __launch_bounds__(256) void node_forward(
    const float* __restrict__ Xin, const float* __restrict__ agg,
    const float* __restrict__ Wlin, const float* __restrict__ Wgate,
    const float* __restrict__ Wsk0, const float* __restrict__ WskL,
    const int* __restrict__ Z,
    float* __restrict__ Xout, float* __restrict__ s0pre, float* __restrict__ gates,
    int NN)
{
  __shared__ float s_agg[8][512];
  __shared__ float s_x[8][512];
  int tid=threadIdx.x;
  int n0=blockIdx.x*8;
  for(int i=tid;i<8*512;i+=256){
    int ln=i>>9, o=i&511;
    int n=n0+ln;
    float av=0.f,xv=0.f;
    if(n<NN){ av=agg[(size_t)n*512+o]; xv=Xin[(size_t)n*512+o]; }
    s_agg[ln][o]=av; s_x[ln][o]=xv;
  }
  __syncthreads();
  int ln=tid>>5, d=tid&31;
  int n=n0+ln;
  if(n>=NN) return;
  int z=Z[n];
  float out[16];
  #pragma unroll
  for(int k=0;k<16;++k) out[k]=0.f;
  constexpr int NL=(T<2)?4:1;
  #pragma unroll
  for(int l=0;l<NL;++l){
    const int na=2*l+1, ao=AOFF[l];
    for(int c=0;c<32;++c){
      float wl=Wlin[l*1024+c*32+d];
      #pragma unroll
      for(int a=0;a<na;++a) out[ao+a]+=s_agg[ln][c*16+ao+a]*wl;
    }
  }
  { // species skip on l=0
    const float* Ws=Wsk0+(size_t)z*1024;
    float s=0.f;
    for(int c=0;c<32;++c) s+=s_x[ln][c*16]*Ws[c*32+d];
    out[0]+=s;
  }
  if(T==1){
    #pragma unroll
    for(int l=1;l<4;++l){
      const float* Ws=WskL+((size_t)(l-1)*NSP+z)*1024;
      const int na=2*l+1, ao=AOFF[l];
      for(int c=0;c<32;++c){
        float wv=Ws[c*32+d];
        #pragma unroll
        for(int a=0;a<na;++a) out[ao+a]+=s_x[ln][c*16+ao+a]*wv;
      }
    }
  }
  if(T<2){
    #pragma unroll
    for(int i=0;i<3;++i){
      float gp=0.f;
      for(int c=0;c<32;++c) gp+=s_agg[ln][c*16]*Wgate[c*96+i*32+d];
      float g=1.f/(1.f+__expf(-gp));
      gates[(size_t)n*96+i*32+d]=g;
      const int l=i+1, na=2*l+1, ao=AOFF[l];
      #pragma unroll
      for(int a=0;a<na;++a) out[ao+a]*=g;
    }
  }
  float p0=out[0];
  s0pre[(size_t)n*32+d]=p0;
  out[0]=siluf(p0);
  float* xo=Xout+(size_t)n*512+d*16;
  #pragma unroll
  for(int k=0;k<16;++k) xo[k]=(T<2)? out[k] : (k==0?out[0]:0.f);
}

// ---------------------------------------------------------------------------
// Node update backward (data-path grads only).
// ---------------------------------------------------------------------------
template<int T>
__global__ __launch_bounds__(256) void node_backward(
  const float* __restrict__ gXout, const float* __restrict__ Xoutsnap,
  const float* __restrict__ gates, const float* __restrict__ s0pre,
  const float* __restrict__ Wlin, const float* __restrict__ Wgate,
  const float* __restrict__ Wsk0, const float* __restrict__ WskL,
  const int* __restrict__ Z,
  float* __restrict__ gAgg, float* __restrict__ gXnext, int NN)
{
  __shared__ float s_g0[8][32];
  __shared__ float s_gl[8][3][7][32];
  __shared__ float s_ggp[8][3][32];
  int tid=threadIdx.x, ln=tid>>5, d=tid&31;
  int n=blockIdx.x*8+ln;
  bool val=(n<NN);
  float go[16];
  #pragma unroll
  for(int k=0;k<16;++k) go[k]=0.f;
  if(val){
    const float* gp=gXout+(size_t)n*512+d*16;
    #pragma unroll
    for(int k=0;k<16;++k) go[k]=gp[k];
  }
  float g0=0.f;
  if(val) g0=go[0]*silugrad(s0pre[(size_t)n*32+d]);
  s_g0[ln][d]=g0;
  if(T<2){
    const float* xo=Xoutsnap+(size_t)n*512+d*16;
    #pragma unroll
    for(int i=0;i<3;++i){
      float gv= val? fmaxf(gates[(size_t)n*96+i*32+d],1e-20f) : 1.f;
      const int l=i+1, na=2*l+1, ao=AOFF[l];
      float gg=0.f;
      #pragma unroll
      for(int a=0;a<7;++a){
        float gpre=0.f;
        if(a<na){
          gpre=go[ao+a]*gv;
          float pregate= val? xo[ao+a]/gv : 0.f;
          gg += go[ao+a]*pregate;
        }
        s_gl[ln][i][a][d]=gpre;
      }
      s_ggp[ln][i][d]=gg*gv*(1.f-gv);
    }
  }
  __syncthreads();
  if(!val) return;
  int c=d, z=Z[n];
  float gA0=0.f;
  for(int dd=0;dd<32;++dd) gA0+=s_g0[ln][dd]*Wlin[0*1024+c*32+dd];
  if(T<2){
    #pragma unroll
    for(int i=0;i<3;++i)
      for(int dd=0;dd<32;++dd) gA0+=s_ggp[ln][i][dd]*Wgate[c*96+i*32+dd];
  }
  gAgg[(size_t)n*512+c*16+0]=gA0;
  if(T<2){
    #pragma unroll
    for(int i=0;i<3;++i){
      const int l=i+1, na=2*l+1, ao=AOFF[l];
      #pragma unroll
      for(int a=0;a<na;++a){
        float s=0.f;
        for(int dd=0;dd<32;++dd) s+=s_gl[ln][i][a][dd]*Wlin[l*1024+c*32+dd];
        gAgg[(size_t)n*512+c*16+ao+a]=s;
      }
    }
  }
  float gx0=0.f;
  for(int dd=0;dd<32;++dd) gx0+=s_g0[ln][dd]*Wsk0[(size_t)z*1024+c*32+dd];
  gXnext[(size_t)n*512+c*16+0]=gx0;
  if(T==1){
    #pragma unroll
    for(int i=0;i<3;++i){
      const int l=i+1, na=2*l+1, ao=AOFF[l];
      #pragma unroll
      for(int a=0;a<na;++a){
        float s=0.f;
        for(int dd=0;dd<32;++dd) s+=s_gl[ln][i][a][dd]*WskL[((size_t)i*NSP+z)*1024+c*32+dd];
        gXnext[(size_t)n*512+c*16+ao+a]=s;
      }
    }
  }
}

// ---------------------------------------------------------------------------
// SiLU backward elementwise: out(bf16) = g * silu'(pre)
// ---------------------------------------------------------------------------
__global__ void silu_bwd_kernel(const float* __restrict__ g, const float* __restrict__ pre,
                                __bf16* __restrict__ outh, size_t n){
  size_t i=(size_t)blockIdx.x*256+threadIdx.x;
  if(i<n) outh[i]=(__bf16)(g[i]*silugrad(pre[i]));
}

// ---------------------------------------------------------------------------
// Geometry backward: fold accumulated g_sh / g_rbf into d e_tot / d edge_vec
// ---------------------------------------------------------------------------
__global__ void geom_backward_kernel(const float* __restrict__ ev,
    const float* __restrict__ gshT, const float* __restrict__ grbT,
    float* __restrict__ fij, int E)
{
  int e=blockIdx.x*blockDim.x+threadIdx.x;
  if(e>=E) return;
  float x=ev[e*3],y=ev[e*3+1],z=ev[e*3+2];
  float r=sqrtf(x*x+y*y+z*z);
  float rm=fmaxf(r,1e-9f);
  float ux=x/rm,uy=y/rm,uz=z/rm;
  const float s3=1.7320508075688772f, s5=2.23606797749979f, s15=3.872983346207417f;
  const float s7=2.6457513110645907f, s42=6.480740698407860f, s105=10.246950765959598f, s70=8.366600265340756f;
  const float* g=gshT+(size_t)e*16;
  float gx=0.f,gy=0.f,gz=0.f;
  gy+=s3*g[1]; gz+=s3*g[2]; gx+=s3*g[3];
  gx+=s15*uy*g[4]; gy+=s15*ux*g[4];
  gy+=s15*uz*g[5]; gz+=s15*uy*g[5];
  gz+=3.f*s5*uz*g[6];
  gx+=s15*uz*g[7]; gz+=s15*ux*g[7];
  gx+=s15*ux*g[8]; gy-=s15*uy*g[8];
  gx+=1.5f*s70*ux*uy*g[9]; gy+=0.75f*s70*(ux*ux-uy*uy)*g[9];
  gx+=s105*uy*uz*g[10]; gy+=s105*ux*uz*g[10]; gz+=s105*ux*uy*g[10];
  gy+=0.25f*s42*(5.f*uz*uz-1.f)*g[11]; gz+=2.5f*s42*uy*uz*g[11];
  gz+=0.5f*s7*(15.f*uz*uz-3.f)*g[12];
  gx+=0.25f*s42*(5.f*uz*uz-1.f)*g[13]; gz+=2.5f*s42*ux*uz*g[13];
  gx+=s105*ux*uz*g[14]; gy-=s105*uy*uz*g[14]; gz+=0.5f*s105*(ux*ux-uy*uy)*g[14];
  gx+=0.75f*s70*(ux*ux-uy*uy)*g[15]; gy-=1.5f*s70*ux*uy*g[15];
  float xr=r/RCf;
  float env=0.f,denv=0.f;
  if(xr<1.f){
    float x2=xr*xr,x3=x2*xr,x5=x2*x3,x6=x5*xr,x7=x6*xr,x8=x7*xr;
    env=1.f-28.f*x6+48.f*x7-21.f*x8;
    denv=(-168.f*x5+336.f*x6-168.f*x7)/RCf;
  }
  float S=sqrtf(2.f/RCf);
  float gr=0.f;
  const float* gb=grbT+(size_t)e*16;
  #pragma unroll
  for(int k=1;k<=8;++k){
    float a=(float)k*PIF/RCf;
    float sn=sinf(a*r), cs=cosf(a*r);
    float db = S*((a*cs/rm - sn/(rm*rm))*env + (sn/rm)*denv);
    gr += gb[k-1]*db;
  }
  float ud=ux*gx+uy*gy+uz*gz;
  fij[e*3+0]= ux*gr + (gx-ux*ud)/rm;
  fij[e*3+1]= uy*gr + (gy-uy*ud)/rm;
  fij[e*3+2]= uz*gr + (gz-uz*ud)/rm;
}

// ---------------------------------------------------------------------------
// Host orchestration
// ---------------------------------------------------------------------------
extern "C" void kernel_launch(void* const* d_in, const int* in_sizes, int n_in,
                              void* d_out, int out_size, void* d_ws, size_t ws_size,
                              hipStream_t stream)
{
  (void)n_in; (void)out_size; (void)ws_size;
  const float* edge_vec=(const float*)d_in[0];
  const float* W_embed =(const float*)d_in[1];
  const float* W_r1=(const float*)d_in[2];
  const float* W_r2=(const float*)d_in[3];
  const float* W_r3=(const float*)d_in[4];
  const float* W_rw=(const float*)d_in[5];
  const float* W_lin=(const float*)d_in[6];
  const float* W_gate=(const float*)d_in[7];
  const float* W_skip0=(const float*)d_in[8];
  const float* W_skipL=(const float*)d_in[9];
  const float* W_out1=(const float*)d_in[10];
  const float* W_out2=(const float*)d_in[11];
  const int* Z   =(const int*)d_in[12];
  const int* esrc=(const int*)d_in[13];
  const int* edst=(const int*)d_in[14];
  const int E  = in_sizes[0]/3;
  const int NN = in_sizes[12];
  float* out=(float*)d_out;

  // ---- workspace layout (bump allocator, 256B aligned) ----
  char* basep=(char*)d_ws; size_t off=0;
  auto alloc=[&](size_t bytes)->void*{ void* p=basep+off; off=(off+bytes+255)&~(size_t)255; return p; };
  float*  cgbuf=(float*)alloc((size_t)CGTOT*4);
  __bf16* wr1p=(__bf16*)alloc((size_t)3*32*64*2);
  __bf16* wr1T=(__bf16*)alloc((size_t)3*64*16*2);
  __bf16* wr2b=(__bf16*)alloc((size_t)3*4096*2);
  __bf16* wr2T=(__bf16*)alloc((size_t)3*4096*2);
  __bf16* wr3b=(__bf16*)alloc((size_t)3*4096*2);
  __bf16* wr3T=(__bf16*)alloc((size_t)3*4096*2);
  __bf16* wrwb=(__bf16*)alloc((size_t)3*64*736*2);
  __bf16* wrwT=(__bf16*)alloc((size_t)3*736*64*2);
  float*  voutb=(float*)alloc(32*4);
  float*  shbuf=(float*)alloc((size_t)E*16*4);
  __bf16* rbfp=(__bf16*)alloc((size_t)E*32*2);
  __bf16* hb1=(__bf16*)alloc((size_t)E*64*2);
  __bf16* hb2=(__bf16*)alloc((size_t)E*64*2);
  float*  pre1=(float*)alloc((size_t)E*64*4);
  float*  pre2=(float*)alloc((size_t)E*64*4);
  float*  pre3=(float*)alloc((size_t)E*64*4);
  float*  ghf =(float*)alloc((size_t)E*64*4);
  __bf16* wbuf=(__bf16*)alloc((size_t)E*736*2);
  float*  Xs[4];
  for(int i=0;i<4;++i) Xs[i]=(float*)alloc((size_t)NN*512*4);
  float*  agg =(float*)alloc((size_t)NN*512*4);
  float*  gAgg=(float*)alloc((size_t)NN*512*4);
  float*  gX[2]; gX[0]=(float*)alloc((size_t)NN*512*4); gX[1]=(float*)alloc((size_t)NN*512*4);
  float*  s0pre=(float*)alloc((size_t)3*NN*32*4);
  float*  gatesb=(float*)alloc((size_t)2*NN*96*4);
  float*  gshT=(float*)alloc((size_t)E*16*4);
  float*  grbT=(float*)alloc((size_t)E*16*4);

  const int NPL[3]={4,23,4};
  auto rungemm=[&](int fuse,const __bf16* A,const __bf16* B,float* oF,__bf16* oH,float* pF,
                   int M,int N,int K,int ldb){
    int tilesM=(M+15)/16;
    dim3 grid((tilesM + 8*GEMM_MT - 1)/(8*GEMM_MT), N/16);
    switch(fuse){
      case 8:  gemm_kernel<8 ><<<grid,256,0,stream>>>(A,B,oF,oH,pF,M,N,K,ldb); break;
      case 9:  gemm_kernel<9 ><<<grid,256,0,stream>>>(A,B,oF,oH,pF,M,N,K,ldb); break;
      case 11: gemm_kernel<11><<<grid,256,0,stream>>>(A,B,oF,oH,pF,M,N,K,ldb); break;
      case 4:  gemm_kernel<4 ><<<grid,256,0,stream>>>(A,B,oF,oH,pF,M,N,K,ldb); break;
      case 16: gemm_kernel<16><<<grid,256,0,stream>>>(A,B,oF,oH,pF,M,N,K,ldb); break;
    }
  };

  // ---- init: CG, weight conversions, geometry, embedding ----
  cg_init_kernel<<<23,343,0,stream>>>(cgbuf);
  for(int t=0;t<3;++t){
    cvt_pad_rows_kernel<<<(32*64+255)/256,256,0,stream>>>(W_r1+(size_t)t*512, wr1p+(size_t)t*2048, 8,64,32);
    cvt_T_pad_kernel  <<<(64*16+255)/256,256,0,stream>>>(W_r1+(size_t)t*512, wr1T+(size_t)t*1024, 8,64,16);
    cvt_kernel        <<<(4096+255)/256,256,0,stream>>>(W_r2+(size_t)t*4096, wr2b+(size_t)t*4096, 4096);
    cvt_T_pad_kernel  <<<(4096+255)/256,256,0,stream>>>(W_r2+(size_t)t*4096, wr2T+(size_t)t*4096, 64,64,64);
    cvt_kernel        <<<(4096+255)/256,256,0,stream>>>(W_r3+(size_t)t*4096, wr3b+(size_t)t*4096, 4096);
    cvt_T_pad_kernel  <<<(4096+255)/256,256,0,stream>>>(W_r3+(size_t)t*4096, wr3T+(size_t)t*4096, 64,64,64);
    cvt_kernel        <<<(47104+255)/256,256,0,stream>>>(W_rw+(size_t)t*47104, wrwb+(size_t)t*47104, 47104);
    cvt_T_pad_kernel  <<<(47104+255)/256,256,0,stream>>>(W_rw+(size_t)t*47104, wrwT+(size_t)t*47104, 64,736,64);
  }
  vout_kernel<<<1,32,0,stream>>>(W_out1,W_out2,voutb);
  edge_geom_kernel<<<(E+255)/256,256,0,stream>>>(edge_vec, shbuf, rbfp, E);
  hipMemsetAsync(Xs[0],0,(size_t)NN*512*4,stream);
  embed_kernel<<<(NN*32+255)/256,256,0,stream>>>(W_embed,Z,Xs[0],NN);

  // ---- forward layers ----
  int tpb=(E+7)/8, nb=(NN+7)/8;
  for(int t=0;t<3;++t){
    int NP=NPL[t];
    hipMemsetAsync(agg,0,(size_t)NN*512*4,stream);
    rungemm(9, rbfp, wr1p+(size_t)t*2048, nullptr, hb1, nullptr, E,64,32,64);
    rungemm(9, hb1,  wr2b+(size_t)t*4096, nullptr, hb2, nullptr, E,64,64,64);
    rungemm(9, hb2,  wr3b+(size_t)t*4096, nullptr, hb1, nullptr, E,64,64,64);
    rungemm(8, hb1,  wrwb+(size_t)t*47104, nullptr, wbuf, nullptr, E, NP*32, 64, 736);
    switch(t){
      case 0: tp_forward<0><<<tpb,256,0,stream>>>(Xs[0],shbuf,wbuf,cgbuf,esrc,edst,agg,E); break;
      case 1: tp_forward<1><<<tpb,256,0,stream>>>(Xs[1],shbuf,wbuf,cgbuf,esrc,edst,agg,E); break;
      case 2: tp_forward<2><<<tpb,256,0,stream>>>(Xs[2],shbuf,wbuf,cgbuf,esrc,edst,agg,E); break;
    }
    const float* Wl = W_lin  + (size_t)t*4096;
    const float* Wg = (t<2)? (W_gate + (size_t)t*3072) : W_gate;
    const float* Ws0= W_skip0+ (size_t)t*NSP*1024;
    float* sp = s0pre + (size_t)t*NN*32;
    float* gt = gatesb + (size_t)((t<2)?t:0)*NN*96;
    switch(t){
      case 0: node_forward<0><<<nb,256,0,stream>>>(Xs[0],agg,Wl,Wg,Ws0,W_skipL,Z,Xs[1],sp,gt,NN); break;
      case 1: node_forward<1><<<nb,256,0,stream>>>(Xs[1],agg,Wl,Wg,Ws0,W_skipL,Z,Xs[2],sp,gt,NN); break;
      case 2: node_forward<2><<<nb,256,0,stream>>>(Xs[2],agg,Wl,Wg,Ws0,W_skipL,Z,Xs[3],sp,gt,NN); break;
    }
  }

  // ---- energy head ----
  hipMemsetAsync(out,0,4,stream);
  head_kernel<<<(NN+255)/256,256,0,stream>>>(Xs[3],voutb,out,NN);

  // ---- backward ----
  hipMemsetAsync(gshT,0,(size_t)E*16*4,stream);
  hipMemsetAsync(grbT,0,(size_t)E*16*4,stream);
  hipMemsetAsync(gX[0],0,(size_t)NN*512*4,stream);
  init_head_grad_kernel<<<(NN*32+255)/256,256,0,stream>>>(voutb,gX[0],NN);
  int cur=0;
  for(int t=2;t>=0;--t){
    int NP=NPL[t];
    int nxt=1-cur;
    hipMemsetAsync(gAgg,0,(size_t)NN*512*4,stream);
    hipMemsetAsync(gX[nxt],0,(size_t)NN*512*4,stream);
    const float* Wl = W_lin  + (size_t)t*4096;
    const float* Wg = (t<2)? (W_gate + (size_t)t*3072) : W_gate;
    const float* Ws0= W_skip0+ (size_t)t*NSP*1024;
    const float* sp = s0pre + (size_t)t*NN*32;
    const float* gt = gatesb + (size_t)((t<2)?t:0)*NN*96;
    switch(t){
      case 2: node_backward<2><<<nb,256,0,stream>>>(gX[cur],Xs[3],gt,sp,Wl,Wg,Ws0,W_skipL,Z,gAgg,gX[nxt],NN); break;
      case 1: node_backward<1><<<nb,256,0,stream>>>(gX[cur],Xs[2],gt,sp,Wl,Wg,Ws0,W_skipL,Z,gAgg,gX[nxt],NN); break;
      case 0: node_backward<0><<<nb,256,0,stream>>>(gX[cur],Xs[1],gt,sp,Wl,Wg,Ws0,W_skipL,Z,gAgg,gX[nxt],NN); break;
    }
    // recompute radial MLP with pre-activations (for silu') and w
    rungemm(11, rbfp, wr1p+(size_t)t*2048,  nullptr, hb1, pre1, E,64,32,64);
    rungemm(11, hb1,  wr2b+(size_t)t*4096,  nullptr, hb2, pre2, E,64,64,64);
    rungemm(11, hb2,  wr3b+(size_t)t*4096,  nullptr, hb1, pre3, E,64,64,64);
    rungemm(8,  hb1,  wrwb+(size_t)t*47104, nullptr, wbuf, nullptr, E, NP*32, 64, 736);
    switch(t){
      case 2: tp_backward<2><<<tpb,256,0,stream>>>(Xs[2],shbuf,wbuf,cgbuf,esrc,edst,gAgg,wbuf,gX[nxt],gshT,E); break;
      case 1: tp_backward<1><<<tpb,256,0,stream>>>(Xs[1],shbuf,wbuf,cgbuf,esrc,edst,gAgg,wbuf,gX[nxt],gshT,E); break;
      case 0: tp_backward<0><<<tpb,256,0,stream>>>(Xs[0],shbuf,wbuf,cgbuf,esrc,edst,gAgg,wbuf,gX[nxt],gshT,E); break;
    }
    // g_w -> g_rbf chain (all WMMA GEMMs with transposed weights)
    int ne=(int)(((size_t)E*64+255)/256);
    rungemm(4, wbuf, wrwT+(size_t)t*47104, ghf, nullptr, nullptr, E,64,NP*32,64);
    silu_bwd_kernel<<<ne,256,0,stream>>>(ghf,pre3,hb2,(size_t)E*64);
    rungemm(4, hb2, wr3T+(size_t)t*4096, ghf, nullptr, nullptr, E,64,64,64);
    silu_bwd_kernel<<<ne,256,0,stream>>>(ghf,pre2,hb1,(size_t)E*64);
    rungemm(4, hb1, wr2T+(size_t)t*4096, ghf, nullptr, nullptr, E,64,64,64);
    silu_bwd_kernel<<<ne,256,0,stream>>>(ghf,pre1,hb2,(size_t)E*64);
    rungemm(16, hb2, wr1T+(size_t)t*1024, grbT, nullptr, nullptr, E,16,64,16);
    cur=nxt;
  }
  geom_backward_kernel<<<(E+255)/256,256,0,stream>>>(edge_vec,gshT,grbT,out+1,E);
}
